// TrajectoryNet_82841329205766
// MI455X (gfx1250) — matrix operand order repaired and checked by reading, not compiled, not run on hardware
//
#include <hip/hip_runtime.h>

// ---------------------------------------------------------------------------
// Types for CDNA5 WMMA
// ---------------------------------------------------------------------------
typedef __attribute__((ext_vector_type(16))) _Float16 v16h;
typedef __attribute__((ext_vector_type(8)))  float    v8f;

__device__ __forceinline__ v8f wmma_f16(v16h a, v16h b, v8f c) {
  // v_wmma_f32_16x16x32_f16  (neg_a, A, neg_b, B, c_mod, C, reuse_a, reuse_b)
  return __builtin_amdgcn_wmma_f32_16x16x32_f16(false, a, false, b, (short)0, c,
                                                false, false);
}

#define ACT_NONE 0
#define ACT_SELU 1
#define ACT_GELU 2

#define KS_LD 72    // padded LDS row stride (halves) for K tile  (144B)
#define VT_LD 264   // padded LDS row stride (halves) for V^T     (528B)

__device__ __forceinline__ float act_selu(float x) {
  const float a = 1.6732632423543772f, s = 1.0507009873554805f;
  return x > 0.f ? s * x : s * a * (__expf(x) - 1.f);
}
__device__ __forceinline__ float act_gelu(float x) {
  return 0.5f * x * (1.f + erff(x * 0.70710678118654752f));
}

// ---------------------------------------------------------------------------
// Weight repack: W[K,N] f32 (row major)  ->  Wt[Np,Kp] f16 (transposed, padded)
// Np is a multiple of 32 so the GEMM's 16x32 wave tile needs no guards.
// ---------------------------------------------------------------------------
__global__ void k_pack_w(const float* __restrict__ W, _Float16* __restrict__ Wt,
                         int K, int N, int Kp, int Np) {
  int idx = blockIdx.x * blockDim.x + threadIdx.x;
  int total = Kp * Np;
  if (idx >= total) return;
  int n = idx / Kp, k = idx % Kp;
  float v = (n < N && k < K) ? W[(size_t)k * N + n] : 0.f;
  Wt[idx] = (_Float16)v;
}

// ---------------------------------------------------------------------------
// Generic WMMA GEMM: Y[M,N] = Res + act( X[M,K] @ W[K,N] + bias )
//   X row-major f32, row stride K (K multiple of 32, caller pads X).
//   Wt transposed padded f16 [Np,Kp=K], Np multiple of 32.
//   Each wave computes a 16x32 tile (two 16x16 WMMA tiles sharing one A frag).
//   rowMode 1: A-row r maps to source row (r>>11)*256 + (r&255)   (B,E,T bcast)
//   grid = (ceil(Np/128), M/16), block = 128 (4 waves)
// ---------------------------------------------------------------------------
__global__ void k_gemm(const float* __restrict__ X, const _Float16* __restrict__ Wt,
                       const float* __restrict__ bias, const float* __restrict__ Res,
                       float* __restrict__ Y, int M, int N, int K, int Np,
                       int act, int rowMode) {
  int l    = threadIdx.x & 31;
  int wv   = threadIdx.x >> 5;
  int col0 = (blockIdx.x * 4 + wv) * 32;
  if (col0 >= Np) return;                       // wave-uniform
  int row0 = blockIdx.y * 16;
  int half = l >> 4, l15 = l & 15;

  int arow = row0 + l15;
  if (rowMode == 1) arow = ((arow >> 11) << 8) | (arow & 255);
  const float*    xr  = X  + (size_t)arow * K;
  const _Float16* wr0 = Wt + (size_t)(col0 + l15) * K;
  const _Float16* wr1 = Wt + (size_t)(col0 + 16 + l15) * K;

  v8f acc0 = {}, acc1 = {};
  for (int k0 = 0; k0 < K; k0 += 32) {
    // ---- A fragment (f32 -> f16):
    // lanes 0-15: k = k0+0..15 ; lanes 16-31: k = k0+16..31
    const float4* p0 = (const float4*)(xr + k0 + half * 16);
    const float4* p1 = (const float4*)(xr + k0 + half * 16 + 8);
    float4 q0 = p0[0], q1 = p0[1];
    float4 r0 = p1[0], r1 = p1[1];
    // ---- B fragments: 16 contiguous f16 per lane (32B vector loads)
    v16h b0 = *(const v16h*)(wr0 + k0 + half * 16);
    v16h b1 = *(const v16h*)(wr1 + k0 + half * 16);
    if (k0 + 32 < K) __builtin_prefetch(xr + k0 + 32, 0, 1);
    v16h a;
    a[0] = (_Float16)q0.x;  a[1] = (_Float16)q0.y;
    a[2] = (_Float16)q0.z;  a[3] = (_Float16)q0.w;
    a[4] = (_Float16)q1.x;  a[5] = (_Float16)q1.y;
    a[6] = (_Float16)q1.z;  a[7] = (_Float16)q1.w;
    a[8]  = (_Float16)r0.x; a[9]  = (_Float16)r0.y;
    a[10] = (_Float16)r0.z; a[11] = (_Float16)r0.w;
    a[12] = (_Float16)r1.x; a[13] = (_Float16)r1.y;
    a[14] = (_Float16)r1.z; a[15] = (_Float16)r1.w;
    acc0 = wmma_f16(a, b0, acc0);
    acc1 = wmma_f16(a, b1, acc1);
  }

  int n0 = col0 + l15;
  if (n0 < N) {
    float bn = bias ? bias[n0] : 0.f;
#pragma unroll
    for (int i = 0; i < 8; i++) {
      int r = row0 + half * 8 + i;
      float v = acc0[i] + bn;
      if (act == ACT_SELU) v = act_selu(v);
      else if (act == ACT_GELU) v = act_gelu(v);
      if (Res) v += Res[(size_t)r * N + n0];
      Y[(size_t)r * N + n0] = v;
    }
  }
  int n1 = col0 + 16 + l15;
  if (n1 < N) {
    float bn = bias ? bias[n1] : 0.f;
#pragma unroll
    for (int i = 0; i < 8; i++) {
      int r = row0 + half * 8 + i;
      float v = acc1[i] + bn;
      if (act == ACT_SELU) v = act_selu(v);
      else if (act == ACT_GELU) v = act_gelu(v);
      if (Res) v += Res[(size_t)r * N + n1];
      Y[(size_t)r * N + n1] = v;
    }
  }
}

// ---------------------------------------------------------------------------
// LayerNorm (population variance, eps 1e-5): one block per row, 128 threads
// ---------------------------------------------------------------------------
__global__ void k_layernorm(const float* __restrict__ X, const float* __restrict__ g,
                            const float* __restrict__ b, float* __restrict__ Y, int D) {
  int row = blockIdx.x, tid = threadIdx.x;
  const float* xr = X + (size_t)row * D;
  float s = 0.f, q = 0.f;
  for (int i = tid; i < D; i += 128) { float v = xr[i]; s += v; q += v * v; }
  for (int off = 16; off; off >>= 1) { s += __shfl_xor(s, off); q += __shfl_xor(q, off); }
  __shared__ float ss[4], sq[4];
  int w = tid >> 5;
  if ((tid & 31) == 0) { ss[w] = s; sq[w] = q; }
  __syncthreads();
  s = ss[0] + ss[1] + ss[2] + ss[3];
  q = sq[0] + sq[1] + sq[2] + sq[3];
  float mean = s / D;
  float var  = q / D - mean * mean;
  float rstd = rsqrtf(var + 1e-5f);
  float* yr = Y + (size_t)row * D;
  for (int i = tid; i < D; i += 128) yr[i] = (xr[i] - mean) * rstd * g[i] + b[i];
}

// ---------------------------------------------------------------------------
// Fused attention, one (head h, seq n) per workgroup. Flash-style online
// softmax, WMMA for Q.K^T and P.V. S=256, dh=64, d_model=512.
// QKV: [64*256, 1536] f32 (q|k|v).  O: [64*256, 512] f32.
// LDS: K [256][KS_LD] f16 + V^T [64][VT_LD] f16 + 4 x P-bounce [16][32] f16
// ---------------------------------------------------------------------------
__global__ void k_attn(const float* __restrict__ QKV, const int* __restrict__ pad,
                       float* __restrict__ O) {
  extern __shared__ char smem[];
  _Float16* Ks = (_Float16*)smem;          // [256][KS_LD]  row-major keys
  _Float16* Vt = Ks + 256 * KS_LD;         // [64][VT_LD]   transposed values
  _Float16* Pb = Vt + 64 * VT_LD;          // 4 waves x [16][32]
  int h = blockIdx.x, n = blockIdx.y;
  int tid = threadIdx.x;

  for (int i = tid; i < 256 * 64; i += 128) {
    int t = i >> 6, d = i & 63;
    const float* base = QKV + ((size_t)n * 256 + t) * 1536 + h * 64 + d;
    Ks[t * KS_LD + d] = (_Float16)base[512];
    Vt[d * VT_LD + t] = (_Float16)base[1024];
  }
  __syncthreads();

  int wv = tid >> 5, l = tid & 31, half = l >> 4, l15 = l & 15;
  _Float16* Pw = Pb + wv * 16 * 32;
  const int* padn = pad + n * 256;

  for (int qb = wv; qb < 16; qb += 4) {
    // Q fragments (16 rows x 64 k, two k-chunks of 32)
    int qrow = n * 256 + qb * 16 + l15;
    const float* qp = QKV + (size_t)qrow * 1536 + h * 64;
    v16h aq[2];
#pragma unroll
    for (int c = 0; c < 2; c++) {
      int kc = c * 32;
#pragma unroll
      for (int e = 0; e < 8; e++) {
        aq[c][e]     = (_Float16)qp[kc + half * 16 + e];
        aq[c][8 + e] = (_Float16)qp[kc + half * 16 + 8 + e];
      }
    }
    float m[8], lsum[8];
    v8f o0 = {}, o1 = {}, o2 = {}, o3 = {};
#pragma unroll
    for (int i = 0; i < 8; i++) { m[i] = -3.0e38f; lsum[i] = 0.f; }

    for (int jj = 0; jj < 8; jj++) {      // 32-key super-tiles
      int kb = jj * 32;
      float p0[8], p1[8];
#pragma unroll
      for (int sub = 0; sub < 2; sub++) {
        int kt = kb + sub * 16;
        const _Float16* kr = Ks + (size_t)(kt + l15) * KS_LD;
        v16h b0 = *(const v16h*)(kr + half * 16);
        v16h b1 = *(const v16h*)(kr + 32 + half * 16);
        v8f sa = {};
        sa = wmma_f16(aq[0], b0, sa);
        sa = wmma_f16(aq[1], b1, sa);
        bool msk = padn[kt + l15] != 0;
        float* pd = sub ? p1 : p0;
#pragma unroll
        for (int i = 0; i < 8; i++) pd[i] = msk ? -1e9f : sa[i] * 0.125f;
      }
      // online softmax over this super-tile (rows live across 16-lane groups)
      float tm[8], al[8], ps[8];
#pragma unroll
      for (int i = 0; i < 8; i++) {
        tm[i] = fmaxf(p0[i], p1[i]);
        for (int off = 1; off < 16; off <<= 1)
          tm[i] = fmaxf(tm[i], __shfl_xor(tm[i], off));
        float nm = fmaxf(m[i], tm[i]);
        al[i] = __expf(m[i] - nm);
        p0[i] = __expf(p0[i] - nm);
        p1[i] = __expf(p1[i] - nm);
        m[i]  = nm;
        ps[i] = p0[i] + p1[i];
        for (int off = 1; off < 16; off <<= 1) ps[i] += __shfl_xor(ps[i], off);
        lsum[i] = lsum[i] * al[i] + ps[i];
        o0[i] *= al[i]; o1[i] *= al[i]; o2[i] *= al[i]; o3[i] *= al[i];
      }
      // bounce P (C-layout) through LDS to build A-layout fragment
#pragma unroll
      for (int i = 0; i < 8; i++) {
        Pw[(half * 8 + i) * 32 + l15]      = (_Float16)p0[i];
        Pw[(half * 8 + i) * 32 + 16 + l15] = (_Float16)p1[i];
      }
      v16h ap;
      const _Float16* pr = Pw + l15 * 32;
#pragma unroll
      for (int e = 0; e < 8; e++) {
        ap[e]     = pr[half * 16 + e];
        ap[8 + e] = pr[half * 16 + 8 + e];
      }
      // P(16x32) @ V(32x64): four 16-wide output tiles, contiguous V^T loads
#pragma unroll
      for (int t4 = 0; t4 < 4; t4++) {
        const _Float16* vrow = Vt + (size_t)(t4 * 16 + l15) * VT_LD + kb + half * 16;
        v16h bv = *(const v16h*)vrow;
        v8f& ot = t4 == 0 ? o0 : t4 == 1 ? o1 : t4 == 2 ? o2 : o3;
        ot = wmma_f16(ap, bv, ot);
      }
    }
    // normalize and write O
#pragma unroll
    for (int i = 0; i < 8; i++) {
      float inv = 1.f / lsum[i];
      int r = n * 256 + qb * 16 + half * 8 + i;
      float* orow = O + (size_t)r * 512 + h * 64;
      orow[l15]      = o0[i] * inv;
      orow[16 + l15] = o1[i] * inv;
      orow[32 + l15] = o2[i] * inv;
      orow[48 + l15] = o3[i] * inv;
    }
  }
}

// ---------------------------------------------------------------------------
// Glue kernels
// ---------------------------------------------------------------------------
__global__ void k_mask(const int* __restrict__ ebc, const int* __restrict__ rt,
                       int* __restrict__ pad) {
  int idx = blockIdx.x * blockDim.x + threadIdx.x;   // 64*256
  if (idx >= 64 * 256) return;
  int n = idx >> 8, t = idx & 255;
  int b = n >> 3, e = n & 7;
  bool bc = ebc[((b * 256 + t) * 8 + e)] != 0;       // [B,T,E,1]
  bool valid = bc && (t < rt[b]);
  pad[idx] = valid ? 0 : 1;
}

__global__ void k_snapx(const float* __restrict__ drone, const float* __restrict__ radio,
                        float* __restrict__ X) {       // [2048, 288] zero-padded
  int r = blockIdx.x;
  for (int c = threadIdx.x; c < 288; c += 128) {
    float v = 0.f;
    if (c < 8) v = drone[r * 8 + c];
    else if (c < 266) v = radio[r * 258 + (c - 8)];
    X[(size_t)r * 288 + c] = v;
  }
}

__global__ void k_gather(const float* __restrict__ tf_out, const int* __restrict__ pad,
                         const int* __restrict__ enb, const int* __restrict__ rt,
                         float* __restrict__ temb) {   // [64, 256]
  int n = blockIdx.x, tid = threadIdx.x;               // 256 threads
  int b = n >> 3, e = n & 7;
  __shared__ int sidx;
  if (tid == 0) sidx = -1;
  __syncthreads();
  if (!pad[n * 256 + tid]) atomicMax(&sidx, tid);
  __syncthreads();
  int li = sidx;
  bool tracked = enb[((b * 256 + (rt[b] - 1)) * 8 + e)] > 0;
  bool ok = tracked && (li >= 0);
  temb[n * 256 + tid] = ok ? tf_out[((size_t)n * 256 + li) * 256 + tid] : 0.f;
}

__global__ void k_ti(const float* __restrict__ temb, float* __restrict__ ti) {
  int r = blockIdx.x;                                  // [16384, 288] padded
  int n = r >> 8, t = r & 255;
  for (int c = threadIdx.x; c < 288; c += 128) {
    float v = 0.f;
    if (c < 256) v = temb[n * 256 + c];
    else if (c == 256) v = (float)t * (1.f / 255.f);
    ti[(size_t)r * 288 + c] = v;
  }
}

__global__ void k_transpose(const float* __restrict__ in, float* __restrict__ out) {
  int idx = blockIdx.x * blockDim.x + threadIdx.x;     // [B,E,T,10] -> [B,T,E,10]
  if (idx >= 8 * 8 * 256 * 10) return;
  int c = idx % 10, t3 = idx / 10;
  int t = t3 % 256, e = (t3 / 256) % 8, b = t3 / (256 * 8);
  out[((b * 256 + t) * 8 + e) * 10 + c] = in[idx];
}

// ---------------------------------------------------------------------------
// Host orchestration
// ---------------------------------------------------------------------------
extern "C" void kernel_launch(void* const* d_in, const int* in_sizes, int n_in,
                              void* d_out, int out_size, void* d_ws, size_t ws_size,
                              hipStream_t stream) {
  (void)in_sizes; (void)n_in; (void)out_size; (void)ws_size;
  const float* drone = (const float*)d_in[0];
  const float* radio = (const float*)d_in[1];
  const int*   ebc   = (const int*)d_in[2];
  const int*   enb   = (const int*)d_in[3];
  const int*   rt    = (const int*)d_in[4];
  // param leaves (jax sorted-key tree flatten), 0-based from d_in[5]
  const float* const* prm = (const float* const*)(d_in + 5);

  char* ws = (char*)d_ws;
  size_t off = 0;
  auto alloc = [&](size_t bytes) -> void* {
    void* p = ws + off;
    off = (off + bytes + 255) & ~(size_t)255;
    return p;
  };
  auto packW = [&](const float* W, int K, int N, int Kp, int Np) -> _Float16* {
    _Float16* dst = (_Float16*)alloc((size_t)Kp * Np * sizeof(_Float16));
    int total = Kp * Np;
    hipLaunchKernelGGL(k_pack_w, dim3((total + 255) / 256), dim3(256), 0, stream,
                       W, dst, K, N, Kp, Np);
    return dst;
  };
  auto gemm = [&](const float* X, const _Float16* Wt, const float* bias,
                  const float* Res, float* Y, int M, int N, int K, int Np,
                  int act, int rowMode) {
    dim3 g((Np + 127) / 128, M / 16);
    hipLaunchKernelGGL(k_gemm, g, dim3(128), 0, stream,
                       X, Wt, bias, Res, Y, M, N, K, Np, act, rowMode);
  };
  auto lnorm = [&](const float* X, const float* g, const float* b, float* Y,
                   int M, int D) {
    hipLaunchKernelGGL(k_layernorm, dim3(M), dim3(128), 0, stream, X, g, b, Y, D);
  };

  // ---- pack all weights to transposed padded f16 (Np multiple of 32) ----
  _Float16* s_in  = packW(prm[35], 266, 64, 288, 64);
  _Float16* s_blk[8];
  for (int i = 0; i < 8; i++) s_blk[i] = packW(prm[4 * i + 1], 64, 64, 64, 64);
  _Float16* s_emb = packW(prm[33], 64, 128, 64, 128);
  _Float16* s_out = packW(prm[39], 128, 5, 128, 32);

  _Float16* tfin = packW(prm[89], 128, 512, 128, 512);
  _Float16 *qkvW[4], *outW[4], *ff1W[4], *ff2W[4];
  for (int L = 0; L < 4; L++) {
    int Li = 40 + 12 * L;
    ff1W[L] = packW(prm[Li + 1], 512, 256, 512, 256);
    ff2W[L] = packW(prm[Li + 3], 256, 512, 256, 512);
    outW[L] = packW(prm[Li + 9], 512, 512, 512, 512);
    qkvW[L] = packW(prm[Li + 11], 512, 1536, 512, 1536);
  }
  _Float16* onin = packW(prm[101], 512, 256, 512, 256);
  _Float16* onb[4];
  for (int i = 0; i < 4; i++) onb[i] = packW(prm[93 + 2 * i], 256, 256, 256, 256);
  _Float16* onout = packW(prm[103], 256, 256, 256, 256);

  _Float16* t_in = packW(prm[139], 257, 256, 288, 256);
  _Float16* t_blk[8];
  for (int i = 0; i < 8; i++) t_blk[i] = packW(prm[105 + 4 * i], 256, 256, 256, 256);
  _Float16* t_emb = packW(prm[137], 256, 256, 256, 256);
  _Float16* t_out = packW(prm[143], 256, 10, 256, 32);

  // ---- activations ------------------------------------------------------
  float* snapX = (float*)alloc((size_t)2048 * 288 * 4);
  float* sh1   = (float*)alloc((size_t)2048 * 64 * 4);
  float* sh2   = (float*)alloc((size_t)2048 * 64 * 4);
  float* sln   = (float*)alloc((size_t)2048 * 64 * 4);
  float* se1   = (float*)alloc((size_t)2048 * 128 * 4);
  float* se2   = (float*)alloc((size_t)2048 * 128 * 4);
  int*   pad   = (int*)alloc((size_t)64 * 256 * 4);
  float* temb  = (float*)alloc((size_t)64 * 256 * 4);
  float* x     = (float*)alloc((size_t)16384 * 512 * 4);
  float* x2    = (float*)alloc((size_t)16384 * 512 * 4);
  float* attnO = (float*)alloc((size_t)16384 * 512 * 4);
  float* big   = (float*)alloc((size_t)16384 * 1536 * 4);   // qkv; later re-sliced
  // aliases inside `big` (only used after the encoder is done with it,
  // except slot0 which doubles as the FF hidden buffer inside each layer)
  float* h1    = big;
  float* h2    = big + (size_t)16384 * 256;
  float* lnb   = big + (size_t)16384 * 256 * 2;
  float* tfout = big + (size_t)16384 * 256 * 3;
  float* tibuf = big + (size_t)16384 * 256 * 4;             // [16384,288]
  float* tpred = big + (size_t)16384 * 256 * 4 + (size_t)16384 * 288;

  float* out_ss = (float*)d_out;            // [8,256,5]
  float* out_tp = (float*)d_out + 10240;    // [8,256,8,10]

  // ---- masks ------------------------------------------------------------
  hipLaunchKernelGGL(k_mask, dim3((16384 + 255) / 256), dim3(256), 0, stream,
                     ebc, rt, pad);

  // ---- snapshot embedding net ------------------------------------------
  hipLaunchKernelGGL(k_snapx, dim3(2048), dim3(128), 0, stream, drone, radio, snapX);
  gemm(snapX, s_in, prm[34], nullptr, sh1, 2048, 64, 288, 64, ACT_SELU, 0);
  for (int i = 0; i < 8; i++) {
    lnorm(sh1, prm[4 * i + 3], prm[4 * i + 2], sln, 2048, 64);
    gemm(sln, s_blk[i], prm[4 * i], sh1, sh2, 2048, 64, 64, 64, ACT_SELU, 0);
    float* t = sh1; sh1 = sh2; sh2 = t;
  }
  gemm(sh1, s_emb, prm[32], nullptr, se1, 2048, 128, 64, 128, ACT_NONE, 0);
  lnorm(se1, prm[37], prm[36], se2, 2048, 128);
  gemm(se2, s_out, prm[38], nullptr, out_ss, 2048, 5, 128, 32, ACT_NONE, 0);

  // ---- transformer input (broadcast per-emitter) ------------------------
  gemm(se2, tfin, prm[88], nullptr, x, 16384, 512, 128, 512, ACT_NONE, 1);

  // ---- encoder layers ---------------------------------------------------
  for (int L = 0; L < 4; L++) {
    int Li = 40 + 12 * L;
    gemm(x, qkvW[L], prm[Li + 10], nullptr, big, 16384, 1536, 512, 1536, ACT_NONE, 0);
    size_t smem = (size_t)(256 * KS_LD + 64 * VT_LD + 4 * 16 * 32) * sizeof(_Float16);
    hipLaunchKernelGGL(k_attn, dim3(8, 64), dim3(128), smem, stream, big, pad, attnO);
    gemm(attnO, outW[L], prm[Li + 8], x, x2, 16384, 512, 512, 512, ACT_NONE, 0);
    lnorm(x2, prm[Li + 5], prm[Li + 4], x, 16384, 512);
    gemm(x, ff1W[L], prm[Li], nullptr, h1, 16384, 256, 512, 256, ACT_GELU, 0);
    gemm(h1, ff2W[L], prm[Li + 2], x, x2, 16384, 512, 256, 512, ACT_NONE, 0);
    lnorm(x2, prm[Li + 7], prm[Li + 6], x, 16384, 512);
  }
  lnorm(x, prm[91], prm[90], x2, 16384, 512);   // final encoder norm

  // ---- out_net ----------------------------------------------------------
  gemm(x2, onin, prm[100], nullptr, h1, 16384, 256, 512, 256, ACT_SELU, 0);
  for (int i = 0; i < 4; i++) {
    gemm(h1, onb[i], prm[92 + 2 * i], h1, h2, 16384, 256, 256, 256, ACT_SELU, 0);
    float* t = h1; h1 = h2; h2 = t;
  }
  gemm(h1, onout, prm[102], nullptr, tfout, 16384, 256, 256, 256, ACT_NONE, 0);

  // ---- gather last valid token per (b,e) --------------------------------
  hipLaunchKernelGGL(k_gather, dim3(64), dim3(256), 0, stream,
                     tfout, pad, enb, rt, temb);
  hipLaunchKernelGGL(k_ti, dim3(16384), dim3(128), 0, stream, temb, tibuf);

  // ---- trajectory embedding net ----------------------------------------
  gemm(tibuf, t_in, prm[138], nullptr, h1, 16384, 256, 288, 256, ACT_SELU, 0);
  for (int i = 0; i < 8; i++) {
    int bi = 104 + 4 * i;
    lnorm(h1, prm[bi + 3], prm[bi + 2], lnb, 16384, 256);
    gemm(lnb, t_blk[i], prm[bi], h1, h2, 16384, 256, 256, 256, ACT_SELU, 0);
    float* t = h1; h1 = h2; h2 = t;
  }
  gemm(h1, t_emb, prm[136], nullptr, h2, 16384, 256, 256, 256, ACT_NONE, 0);
  lnorm(h2, prm[141], prm[140], h1, 16384, 256);
  gemm(h1, t_out, prm[142], nullptr, tpred, 16384, 10, 256, 32, ACT_NONE, 0);

  hipLaunchKernelGGL(k_transpose, dim3((163840 + 255) / 256), dim3(256), 0, stream,
                     tpred, out_tp);
}
